// MobiusGRUCell_52432960749610
// MI455X (gfx1250) — matrix operand order repaired and checked by reading, not compiled, not run on hardware
//
#include <hip/hip_runtime.h>
#include <hip/hip_bf16.h>
#include <stdint.h>

#define D       256
#define ROWS    32          // rows per workgroup
#define THREADS 64          // 2 waves of 32

#define MINN 1e-15f
#define MAXN (1.0f - 1e-5f)

// pre-swizzled B-fragment buffer: per matrix 16(nt) x 8(kc) x 32(lane) x 8(v) u32
#define PREW_U32_PER_MAT (16 * 8 * 32 * 8)          // 32768 u32 = 128 KB
#define PREW_BYTES       (5u * PREW_U32_PER_MAT * 4u)  // 640 KB

typedef __bf16   v16bf __attribute__((ext_vector_type(16)));
typedef float    v8f   __attribute__((ext_vector_type(8)));
typedef uint32_t v8u   __attribute__((ext_vector_type(8)));

// ---- f32 -> bf16 (round-to-nearest-even) ----
__device__ __forceinline__ uint32_t f2bf(float f) {
  uint32_t u = __builtin_bit_cast(uint32_t, f);
  return (u + 0x7fffu + ((u >> 16) & 1u)) >> 16;
}
__device__ __forceinline__ uint32_t packbf(float lo, float hi) {
  return f2bf(lo) | (f2bf(hi) << 16);
}

// ---- pair (lanes 2r, 2r+1) reduction ----
__device__ __forceinline__ float psum(float v) { return v + __shfl_xor(v, 1, 32); }

// =======================================================================
// Weight pre-swizzle: f32 [256][256] -> bf16-pair fragments laid out so a
// lane's 8 dwords for one (nt,kc) fragment are contiguous (2x b128 loads).
// B frag layout (16-bit 32x16): lane l: g=l>>4, n=l&15; dword v holds
// K = kc*32 + g*16 + 2v (lo) and K+1 (hi), column nt*16+n.
// =======================================================================
__global__ void prep_weights_kernel(const float* __restrict__ w0, const float* __restrict__ w1,
                                    const float* __restrict__ w2, const float* __restrict__ w3,
                                    const float* __restrict__ w4, uint32_t* __restrict__ outb) {
  const float* Ws[5] = {w0, w1, w2, w3, w4};
  const int mat = blockIdx.y;
  const float* W = Ws[mat];
  uint32_t* o = outb + (size_t)mat * PREW_U32_PER_MAT;
  int i = blockIdx.x * blockDim.x + threadIdx.x;   // grid covers 32768 per matrix
  if (i >= PREW_U32_PER_MAT) return;
  const int v    = i & 7;
  const int lane = (i >> 3) & 31;
  const int kc   = (i >> 8) & 7;
  const int nt   = i >> 11;
  const int g = lane >> 4, n = lane & 15;
  const int K = kc * 32 + g * 16 + 2 * v;
  const int col = nt * 16 + n;
  o[i] = packbf(W[(size_t)K * D + col], W[(size_t)(K + 1) * D + col]);
}

// =======================================================================
// A-fragment builders (16-bit 16x32 layout)
// =======================================================================
__device__ __forceinline__ void build_a_f32(const float* __restrict__ row, int g, v16bf a[8]) {
  #pragma unroll
  for (int kc = 0; kc < 8; ++kc) {
    v8u au;
    #pragma unroll
    for (int v = 0; v < 8; ++v) {
      int K = kc * 32 + g * 16 + 2 * v;
      au[v] = packbf(row[K], row[K + 1]);
    }
    a[kc] = __builtin_bit_cast(v16bf, au);
  }
}

__device__ __forceinline__ void build_a_bf16(const uint32_t* __restrict__ rowp, int g, v16bf a[8]) {
  #pragma unroll
  for (int kc = 0; kc < 8; ++kc) {
    v8u au;
    #pragma unroll
    for (int v = 0; v < 8; ++v) {
      int K = kc * 32 + g * 16 + 2 * v;
      au[v] = rowp[K >> 1];
    }
    a[kc] = __builtin_bit_cast(v16bf, au);
  }
}

// Full 16x256 = A(16x256) x W(256x256) GEMM for one wave; tiles scattered to LDS.
// PRE=true: B fragments loaded pre-swizzled from preW; else gathered+converted from W.
template <bool PRE>
__device__ void wmma_gemm(const v16bf a[8], const float* __restrict__ W,
                          const uint32_t* __restrict__ preW,
                          float* lbuf, int wbase, int g, int n, int lane) {
  for (int nt = 0; nt < 16; ++nt) {
    v8f acc;
    #pragma unroll
    for (int i = 0; i < 8; ++i) acc[i] = 0.0f;
    #pragma unroll
    for (int kc = 0; kc < 8; ++kc) {
      v8u bu;
      if (PRE) {
        const uint32_t* F = preW + (((nt * 8 + kc) * 32 + lane) << 3);
        #pragma unroll
        for (int v = 0; v < 8; ++v) bu[v] = F[v];     // 2x global_load_b128
      } else {
        #pragma unroll
        for (int v = 0; v < 8; ++v) {
          int K = kc * 32 + g * 16 + 2 * v;
          const float* p = W + (size_t)K * D + nt * 16 + n;
          bu[v] = packbf(p[0], p[D]);
        }
      }
      v16bf bfrag = __builtin_bit_cast(v16bf, bu);
      acc = __builtin_amdgcn_wmma_f32_16x16x32_bf16(
          false, a[kc], false, bfrag, (short)0, acc, false, false);
    }
    // C/D layout: VGPR j -> M = j (lanes 0-15) / j+8 (lanes 16-31), N = lane&15
    #pragma unroll
    for (int j = 0; j < 8; ++j)
      lbuf[(wbase + g * 8 + j) * D + nt * 16 + n] = acc[j];
  }
}

// =======================================================================
// Row-wise Mobius helpers (lane pair per row, 128-col halves).
// =======================================================================
__device__ __forceinline__ void mob_scale_row(float* mh, float srcn2) {
  float mn2 = 0.0f;
  #pragma unroll 4
  for (int c = 0; c < 128; ++c) mn2 += mh[c] * mh[c];
  mn2 = psum(mn2);
  float mn = sqrtf(fmaxf(mn2, MINN));
  float xn = sqrtf(fmaxf(srcn2, MINN));
  xn = fminf(fmaxf(xn, MINN), MAXN);
  float t = tanhf((mn / xn) * atanhf(xn));
  float s = t / mn;
  if (t > MAXN) s *= MAXN / t;               // project()
  #pragma unroll 4
  for (int c = 0; c < 128; ++c) mh[c] *= s;
}

__device__ __forceinline__ void mob_add_row(const float* xh, float sx,
                                            const float* yh, float* oh) {
  float x2 = 0.0f, y2 = 0.0f, xy = 0.0f;
  #pragma unroll 4
  for (int c = 0; c < 128; ++c) {
    float xv = xh[c], yv = yh[c];
    x2 += xv * xv; y2 += yv * yv; xy += xv * yv;
  }
  x2 = psum(x2); y2 = psum(y2); xy = psum(xy) * sx;
  float a   = 1.0f + 2.0f * xy + y2;
  float b   = 1.0f - x2;
  float den = fmaxf(1.0f + 2.0f * xy + x2 * y2, MINN);
  float n2  = (a * a * x2 + 2.0f * a * b * xy + b * b * y2) / (den * den);
  float n   = sqrtf(fmaxf(n2, MINN));
  float f   = (n > MAXN) ? (MAXN / n) : 1.0f;
  float fa = f * a / den * sx, fb = f * b / den;
  #pragma unroll 4
  for (int c = 0; c < 128; ++c) oh[c] = fa * xh[c] + fb * yh[c];
}

// sigmoid(layernorm(log_map_zero(v))) fused
__device__ __forceinline__ void log_ln_sig(const float* vh, const float* gg,
                                           const float* bb, float* oh) {
  float s1 = 0.0f, s2 = 0.0f;
  #pragma unroll 4
  for (int c = 0; c < 128; ++c) { float v = vh[c]; s1 += v; s2 += v * v; }
  s1 = psum(s1); s2 = psum(s2);
  float n = sqrtf(fmaxf(s2, MINN));
  n = fminf(fmaxf(n, MINN), MAXN);
  float ls  = atanhf(n) / n;
  float mu  = ls * s1 * (1.0f / 256.0f);
  float var = ls * ls * s2 * (1.0f / 256.0f) - mu * mu;
  float inv = rsqrtf(var + 1e-5f);
  #pragma unroll 4
  for (int c = 0; c < 128; ++c) {
    float y = (ls * vh[c] - mu) * inv * gg[c] + bb[c];
    oh[c] = 1.0f / (1.0f + expf(-y));
  }
}

// =======================================================================
template <bool PRE>
__global__ __launch_bounds__(THREADS)
void mobius_gru_kernel(const float* __restrict__ hyp_x, const float* __restrict__ hidden,
                       const float* __restrict__ w_z, const float* __restrict__ w_r,
                       const float* __restrict__ w_h, const float* __restrict__ u_z,
                       const float* __restrict__ u_r,
                       const float* __restrict__ b_z, const float* __restrict__ b_r,
                       const float* __restrict__ b_h,
                       const float* __restrict__ ln_z_g, const float* __restrict__ ln_z_b,
                       const float* __restrict__ ln_r_g, const float* __restrict__ ln_r_b,
                       const uint32_t* __restrict__ preW,
                       float* __restrict__ out) {
  __shared__ float    BUFH[ROWS * D];        // hidden rows (f32)
  __shared__ float    BUF0[ROWS * D];        // gemm out / temps
  __shared__ float    BUF1[ROWS * D];        // gemm out / Ux_r / temps
  __shared__ float    BUFZ[ROWS * D];        // z gate
  __shared__ uint32_t BUFP[ROWS * (D / 2)];  // r (.) h as packed bf16 pairs

  const int tid  = threadIdx.x;
  const int wave = tid >> 5;
  const int lane = tid & 31;
  const int g    = lane >> 4;       // WMMA lane group
  const int m    = lane & 15;       // WMMA row/col within tile
  const int wbase = wave * 16;      // wave's first local row
  const int blkRow = blockIdx.x * ROWS;

  const uint32_t* preWz = preW;
  const uint32_t* preWr = preW + 1 * PREW_U32_PER_MAT;
  const uint32_t* preWh = preW + 2 * PREW_U32_PER_MAT;
  const uint32_t* preUz = preW + 3 * PREW_U32_PER_MAT;
  const uint32_t* preUr = preW + 4 * PREW_U32_PER_MAT;

  // ---- stage hidden rows into LDS (coalesced) ----
  {
    const float4* src = (const float4*)(hidden + (size_t)blkRow * D);
    float4* dst = (float4*)BUFH;
    for (int i = tid; i < ROWS * D / 4; i += THREADS) dst[i] = src[i];
  }
  __syncthreads();

  // ---- per-row identities for nonlinear phases (lane pair per row) ----
  const int rloc = wbase + (lane >> 1);
  const int half = lane & 1;
  const int colb = half * 128;
  const size_t rglob = (size_t)(blkRow + rloc);
  float* Hrow = BUFH + rloc * D + colb;
  float* R0   = BUF0 + rloc * D + colb;
  float* R1   = BUF1 + rloc * D + colb;
  float* RZ   = BUFZ + rloc * D + colb;

  const float* Hfrag = BUFH + (wbase + m) * D;                       // LDS row for A frags
  const float* Xfrag = hyp_x + (size_t)(blkRow + wbase + m) * D;     // global row for A frags

  v16bf aF[8];   // single live A-fragment set (rebuilt before each GEMM)

  // ================= z gate =================
  build_a_f32(Hfrag, g, aF);
  wmma_gemm<PRE>(aF, w_z, preWz, BUF0, wbase, g, m, lane);   // Mz  = h @ w_z
  build_a_f32(Xfrag, g, aF);
  wmma_gemm<PRE>(aF, u_z, preUz, BUF1, wbase, g, m, lane);   // Mxz = x @ u_z

  float hn2 = 0.0f, xn2 = 0.0f;
  {
    const float* Xrow = hyp_x + rglob * D + colb;
    #pragma unroll 4
    for (int c = 0; c < 128; ++c) { hn2 += Hrow[c] * Hrow[c]; xn2 += Xrow[c] * Xrow[c]; }
    hn2 = psum(hn2); xn2 = psum(xn2);
  }

  mob_scale_row(R0, hn2);                  // Wh_z
  mob_scale_row(R1, xn2);                  // Ux_z
  mob_add_row(R0, 1.0f, R1, R0);           // Wh_z (+) Ux_z
  mob_add_row(R0, 1.0f, b_z + colb, R0);   // (+) b_z
  log_ln_sig(R0, ln_z_g + colb, ln_z_b + colb, RZ);   // z -> BUFZ

  // ================= r gate =================
  build_a_f32(Hfrag, g, aF);
  wmma_gemm<PRE>(aF, w_r, preWr, BUF0, wbase, g, m, lane);   // Mr  = h @ w_r
  build_a_f32(Xfrag, g, aF);
  wmma_gemm<PRE>(aF, u_r, preUr, BUF1, wbase, g, m, lane);   // Mxr = x @ u_r

  mob_scale_row(R0, hn2);                  // Wh_r
  mob_scale_row(R1, xn2);                  // Ux_r  (kept in BUF1 for h_tilde!)
  mob_add_row(R0, 1.0f, R1, R0);
  mob_add_row(R0, 1.0f, b_r + colb, R0);
  log_ln_sig(R0, ln_r_g + colb, ln_r_b + colb, R0);   // r -> BUF0

  // ---- r_point_h = mob_pointwise_prod(h, r) -> BUFP (bf16) ----
  float rphn2;
  {
    float vacc = 0.0f;
    #pragma unroll 4
    for (int c = 0; c < 128; ++c) { float v = Hrow[c] * R0[c]; vacc += v * v; }
    float vn2 = psum(vacc);
    float vn  = sqrtf(fmaxf(vn2, MINN));
    float xn  = sqrtf(fmaxf(hn2, MINN));
    xn = fminf(fmaxf(xn, MINN), MAXN);
    float t = tanhf((vn / xn) * atanhf(xn));
    float s = t / vn;
    float rn = t;
    if (t > MAXN) { s *= MAXN / t; rn = MAXN; }
    rphn2 = rn * rn;
    uint32_t* P = BUFP + rloc * (D / 2) + (colb >> 1);
    #pragma unroll 4
    for (int c = 0; c < 128; c += 2)
      P[c >> 1] = packbf(s * Hrow[c] * R0[c], s * Hrow[c + 1] * R0[c + 1]);
  }

  // ================= candidate =================
  build_a_bf16(BUFP + (wbase + m) * (D / 2), g, aF);
  wmma_gemm<PRE>(aF, w_h, preWh, BUF0, wbase, g, m, lane);   // Mh = rph @ w_h

  mob_scale_row(R0, rphn2);                // Wh
  mob_add_row(R0, 1.0f, R1, R0);           // Wh (+) Ux_r
  mob_add_row(R0, 1.0f, b_h + colb, R0);   // h_tilde
  mob_add_row(Hrow, -1.0f, R0, R1);        // (-h) (+) h_tilde -> BUF1

  // ---- mob_pointwise_prod(minus, z) in place in BUF1 ----
  {
    float mn2 = 0.0f, vacc = 0.0f;
    #pragma unroll 4
    for (int c = 0; c < 128; ++c) {
      float mv = R1[c];
      mn2 += mv * mv;
      float v = mv * RZ[c];
      vacc += v * v;
    }
    mn2 = psum(mn2); vacc = psum(vacc);
    float vn = sqrtf(fmaxf(vacc, MINN));
    float xn = sqrtf(fmaxf(mn2, MINN));
    xn = fminf(fmaxf(xn, MINN), MAXN);
    float t = tanhf((vn / xn) * atanhf(xn));
    float s = t / vn;
    if (t > MAXN) s *= MAXN / t;
    #pragma unroll 4
    for (int c = 0; c < 128; ++c) R1[c] = s * R1[c] * RZ[c];
  }

  mob_add_row(Hrow, 1.0f, R1, R0);         // new_h -> BUF0

  __syncthreads();
  // ---- coalesced store ----
  {
    float4* dst = (float4*)(out + (size_t)blkRow * D);
    const float4* src = (const float4*)BUF0;
    for (int i = tid; i < ROWS * D / 4; i += THREADS) dst[i] = src[i];
  }
}

extern "C" void kernel_launch(void* const* d_in, const int* in_sizes, int n_in,
                              void* d_out, int out_size, void* d_ws, size_t ws_size,
                              hipStream_t stream) {
  (void)n_in; (void)out_size;
  const float* hyp_x  = (const float*)d_in[0];
  const float* hidden = (const float*)d_in[1];
  const float* w_z    = (const float*)d_in[2];
  const float* w_r    = (const float*)d_in[3];
  const float* w_h    = (const float*)d_in[4];
  const float* u_z    = (const float*)d_in[5];
  const float* u_r    = (const float*)d_in[6];
  // d_in[7] = u_h: unused (reference faithfully reuses u_r for h_tilde)
  const float* b_z    = (const float*)d_in[8];
  const float* b_r    = (const float*)d_in[9];
  const float* b_h    = (const float*)d_in[10];
  const float* ln_z_g = (const float*)d_in[11];
  const float* ln_z_b = (const float*)d_in[12];
  const float* ln_r_g = (const float*)d_in[13];
  const float* ln_r_b = (const float*)d_in[14];

  const int B = in_sizes[0] / D;           // 131072
  dim3 grid(B / ROWS), block(THREADS);

  const bool pre = (d_ws != nullptr) && (ws_size >= (size_t)PREW_BYTES);
  if (pre) {
    uint32_t* wsb = (uint32_t*)d_ws;
    dim3 pgrid(PREW_U32_PER_MAT / 256, 5), pblock(256);
    prep_weights_kernel<<<pgrid, pblock, 0, stream>>>(w_z, w_r, w_h, u_z, u_r, wsb);
    mobius_gru_kernel<true><<<grid, block, 0, stream>>>(
        hyp_x, hidden, w_z, w_r, w_h, u_z, u_r,
        b_z, b_r, b_h, ln_z_g, ln_z_b, ln_r_g, ln_r_b, wsb, (float*)d_out);
  } else {
    mobius_gru_kernel<false><<<grid, block, 0, stream>>>(
        hyp_x, hidden, w_z, w_r, w_h, u_z, u_r,
        b_z, b_r, b_h, ln_z_g, ln_z_b, ln_r_g, ln_r_b, (const uint32_t*)d_ws, (float*)d_out);
  }
}
